// Layer_44126493999719
// MI455X (gfx1250) — matrix operand order repaired and checked by reading, not compiled, not run on hardware
//
#include <hip/hip_runtime.h>
#include <hip/hip_bf16.h>

// ---------------- types ----------------
typedef __attribute__((ext_vector_type(16))) __bf16          v16bf;
typedef __attribute__((ext_vector_type(8)))  float           v8f;
typedef __attribute__((ext_vector_type(8)))  unsigned short  us8;
typedef __attribute__((ext_vector_type(4)))  unsigned short  us4;

#define DD 128   // feature dim

__device__ __forceinline__ unsigned short f2bf(float f) {
    unsigned u = __float_as_uint(f);
    unsigned r = u + 0x7FFFu + ((u >> 16) & 1u);   // RNE to bf16
    return (unsigned short)(r >> 16);
}

__device__ __forceinline__ void st4bf(unsigned short* p, float a, float b, float c, float d) {
    us4 v = { f2bf(a), f2bf(b), f2bf(c), f2bf(d) };
    *(us4*)p = v;   // ds_store_b64
}

// ---------------- weight pre-pack: (K x 128) f32 -> per-lane bf16 B-fragments ----------------
// dest index: (((kc*8 + t)*32 + lane)*16 + i),  lane = ((k%32)/16)*16 + (n%16), i = k%16, t = n/16
__global__ __launch_bounds__(256) void pack_w_kernel(const float* __restrict__ W,
                                                     unsigned short* __restrict__ out, int K) {
    int idx = blockIdx.x * 256 + threadIdx.x;
    if (idx >= K * DD) return;
    int k = idx >> 7, n = idx & 127;
    int kc = k >> 5, r = k & 31;
    int lane = ((r >> 4) << 4) | (n & 15);
    int i = r & 15;
    int t = n >> 4;
    out[(size_t)(((kc * 8 + t) * 32 + lane)) * 16 + i] = f2bf(W[idx]);
}

// ---------------- zero workspace region ----------------
__global__ __launch_bounds__(256) void zero_kernel(float4* p, size_t n4) {
    size_t i = (size_t)blockIdx.x * 256 + threadIdx.x;
    if (i < n4) p[i] = make_float4(0.f, 0.f, 0.f, 0.f);
}

// ---------------- fused gather + bf16 WMMA GEMM + column sum/sumsq ----------------
// MODE 0: edge   : A row e = [0.5*(x[row]+x[col]) | edge_attr[e] | u[batch[row]]], K=384
// MODE 1: node1  : A row e = [x[col] | edge_out[e]],                               K=256
// MODE 2: node2  : A row i = [agg[i]/max(cnt,1) | u[batch[i]]],                    K=256
template <int MODE, int KCH>
__global__ __launch_bounds__(128)
void gemm_bn_kernel(const float* __restrict__ x,
                    const int*   __restrict__ row_idx,
                    const int*   __restrict__ col_idx,
                    const float* __restrict__ eattr,
                    const float* __restrict__ u,
                    const int*   __restrict__ batch,
                    const float* __restrict__ eout,
                    const float* __restrict__ agg,
                    const float* __restrict__ cnt,
                    const unsigned short* __restrict__ pB,
                    float* __restrict__ y,
                    float* __restrict__ gsum,
                    float* __restrict__ gsq,
                    int nrows) {
    constexpr int KTOT  = KCH * 32;
    constexpr int PITCH = KTOT + 8;               // ushorts per staged row (pad for LDS banks)
    __shared__ __align__(16) unsigned short sA[4 * 16 * PITCH];
    __shared__ float sSum[DD];
    __shared__ float sSq[DD];

    const int tid  = threadIdx.x;
    const int wave = tid >> 5;
    const int lane = tid & 31;
    sSum[tid] = 0.f;
    sSq[tid]  = 0.f;

    const int rbase = blockIdx.x * 64 + wave * 16;
    unsigned short* wrow = &sA[wave * 16 * PITCH];

    // ---- stage 16 rows of A as bf16 into LDS (zero-fill OOB rows: they contribute 0 to stats) ----
    for (int rr = 0; rr < 16; ++rr) {
        int r = rbase + rr;
        unsigned short* dst = wrow + rr * PITCH;
        if (r < nrows) {
            if constexpr (MODE == 0) {
                int s = row_idx[r], c = col_idx[r];
                int b = batch[s];
                float4 a4 = *(const float4*)(x + (size_t)s * DD + lane * 4);
                float4 b4 = *(const float4*)(x + (size_t)c * DD + lane * 4);
                st4bf(dst + lane * 4, 0.5f * (a4.x + b4.x), 0.5f * (a4.y + b4.y),
                                      0.5f * (a4.z + b4.z), 0.5f * (a4.w + b4.w));
                float4 e4 = *(const float4*)(eattr + (size_t)r * DD + lane * 4);
                st4bf(dst + DD + lane * 4, e4.x, e4.y, e4.z, e4.w);
                float4 u4 = *(const float4*)(u + (size_t)b * DD + lane * 4);
                st4bf(dst + 2 * DD + lane * 4, u4.x, u4.y, u4.z, u4.w);
            } else if constexpr (MODE == 1) {
                int c = col_idx[r];
                float4 a4 = *(const float4*)(x + (size_t)c * DD + lane * 4);
                st4bf(dst + lane * 4, a4.x, a4.y, a4.z, a4.w);
                float4 e4 = *(const float4*)(eout + (size_t)r * DD + lane * 4);
                st4bf(dst + DD + lane * 4, e4.x, e4.y, e4.z, e4.w);
            } else {
                int b = batch[r];
                float inv = 1.f / fmaxf(cnt[r], 1.f);
                float4 a4 = *(const float4*)(agg + (size_t)r * DD + lane * 4);
                st4bf(dst + lane * 4, a4.x * inv, a4.y * inv, a4.z * inv, a4.w * inv);
                float4 u4 = *(const float4*)(u + (size_t)b * DD + lane * 4);
                st4bf(dst + DD + lane * 4, u4.x, u4.y, u4.z, u4.w);
            }
        } else {
            for (int o = lane * 4; o < KTOT; o += 128) st4bf(dst + o, 0.f, 0.f, 0.f, 0.f);
        }
    }
    __syncthreads();

    // ---- WMMA: 16 rows x 128 cols per wave, 8 tiles of 16x16, K chunks of 32 ----
    v8f zero8 = {0.f, 0.f, 0.f, 0.f, 0.f, 0.f, 0.f, 0.f};
    v8f acc[8];
#pragma unroll
    for (int t = 0; t < 8; ++t) acc[t] = zero8;

    const int m  = lane & 15;    // A row / C column within tile (lanes 0-15 & 16-31 both map M=0..15)
    const int kh = lane >> 4;    // K-half selector: contiguous k = 16*kh + s (same order as the B pack)
    const unsigned short* arow = wrow + m * PITCH;
    const v16bf* bfrag = (const v16bf*)pB + (size_t)lane;

#pragma unroll
    for (int kc = 0; kc < KCH; ++kc) {
        if (kc + 1 < KCH)
            __builtin_prefetch((const void*)(bfrag + (size_t)(kc + 1) * 8 * 32), 0, 0);
        union { v16bf v; us8 h[2]; } A;
        const unsigned short* p0 = arow + kc * 32 + kh * 16;
        A.h[0] = *(const us8*)p0;            // K[kh*16 .. +7]   (ds_load_b128)
        A.h[1] = *(const us8*)(p0 + 8);      // K[kh*16+8 .. +7]
#pragma unroll
        for (int t = 0; t < 8; ++t) {
            v16bf B = bfrag[(size_t)(kc * 8 + t) * 32];
            acc[t] = __builtin_amdgcn_wmma_f32_16x16x32_bf16(
                false, A.v, false, B, (short)0, acc[t], false, false);
        }
    }

    // ---- store pre-BN y + column sum/sumsq (LDS reduce, then global atomics) ----
#pragma unroll
    for (int t = 0; t < 8; ++t) {
        float s = 0.f, q = 0.f;
#pragma unroll
        for (int j = 0; j < 8; ++j) { float v = acc[t][j]; s += v; q += v * v; }
        s += __shfl_xor(s, 16, 32);
        q += __shfl_xor(q, 16, 32);
        if (lane < 16) {
            atomicAdd(&sSum[t * 16 + lane], s);
            atomicAdd(&sSq[t * 16 + lane], q);
        }
#pragma unroll
        for (int j = 0; j < 8; ++j) {
            int r = rbase + kh * 8 + j;
            if (r < nrows) y[(size_t)r * DD + t * 16 + m] = acc[t][j];
        }
    }
    __syncthreads();
    atomicAdd(&gsum[tid], sSum[tid]);
    atomicAdd(&gsq[tid],  sSq[tid]);
}

// ---------------- BN finalize: per-column scale/shift ----------------
__global__ void bn_finalize_kernel(const float* __restrict__ sum, const float* __restrict__ sq,
                                   const float* __restrict__ gamma, const float* __restrict__ beta,
                                   float* __restrict__ scale, float* __restrict__ shift, float invn) {
    int c = threadIdx.x;
    float m = sum[c] * invn;
    float v = fmaxf(sq[c] * invn - m * m, 0.f);
    float sc = gamma[c] * rsqrtf(v + 1e-5f);
    scale[c] = sc;
    shift[c] = beta[c] - m * sc;
}

// ---------------- BN + PReLU in place (edge_out) ----------------
__global__ __launch_bounds__(256) void bn_apply_kernel(float* __restrict__ y,
                                                       const float* __restrict__ scale,
                                                       const float* __restrict__ shift,
                                                       const float* __restrict__ aptr, size_t n4) {
    size_t i = (size_t)blockIdx.x * 256 + threadIdx.x;
    if (i >= n4) return;
    float a = *aptr;
    int c = (int)((i * 4) & (DD - 1));
    float4 v = ((float4*)y)[i];
    v.x = v.x * scale[c + 0] + shift[c + 0]; v.x = v.x >= 0.f ? v.x : a * v.x;
    v.y = v.y * scale[c + 1] + shift[c + 1]; v.y = v.y >= 0.f ? v.y : a * v.y;
    v.z = v.z * scale[c + 2] + shift[c + 2]; v.z = v.z >= 0.f ? v.z : a * v.z;
    v.w = v.w * scale[c + 3] + shift[c + 3]; v.w = v.w >= 0.f ? v.w : a * v.w;
    ((float4*)y)[i] = v;
}

// ---------------- BN + PReLU on y_h, fused scatter-add by edge row ----------------
__global__ __launch_bounds__(256) void h_scatter_kernel(const float* __restrict__ yh,
                                                        const float* __restrict__ scale,
                                                        const float* __restrict__ shift,
                                                        const float* __restrict__ aptr,
                                                        const int* __restrict__ row_idx,
                                                        float* __restrict__ agg,
                                                        float* __restrict__ cnt, int Erows) {
    size_t g = (size_t)blockIdx.x * 256 + threadIdx.x;   // over E*32 float4 groups
    if (g >= (size_t)Erows * 32) return;
    int e = (int)(g >> 5);
    int c = ((int)g & 31) * 4;
    float a = *aptr;
    int r = row_idx[e];
    float4 v = *(const float4*)(yh + (size_t)e * DD + c);
    v.x = v.x * scale[c + 0] + shift[c + 0]; v.x = v.x >= 0.f ? v.x : a * v.x;
    v.y = v.y * scale[c + 1] + shift[c + 1]; v.y = v.y >= 0.f ? v.y : a * v.y;
    v.z = v.z * scale[c + 2] + shift[c + 2]; v.z = v.z >= 0.f ? v.z : a * v.z;
    v.w = v.w * scale[c + 3] + shift[c + 3]; v.w = v.w >= 0.f ? v.w : a * v.w;
    float* dst = agg + (size_t)r * DD + c;
    atomicAdd(dst + 0, v.x);
    atomicAdd(dst + 1, v.y);
    atomicAdd(dst + 2, v.z);
    atomicAdd(dst + 3, v.w);
    if ((g & 31) == 0) atomicAdd(&cnt[r], 1.f);
}

// ---------------- BN + PReLU on y_x in place (-> x_out), fused scatter by graph ----------------
__global__ __launch_bounds__(256) void x_scatter_kernel(float* __restrict__ yx,
                                                        const float* __restrict__ scale,
                                                        const float* __restrict__ shift,
                                                        const float* __restrict__ aptr,
                                                        const int* __restrict__ batch,
                                                        float* __restrict__ gsumG,
                                                        float* __restrict__ gcnt, int Nrows) {
    size_t g = (size_t)blockIdx.x * 256 + threadIdx.x;   // over N*32 float4 groups
    if (g >= (size_t)Nrows * 32) return;
    int i = (int)(g >> 5);
    int c = ((int)g & 31) * 4;
    float a = *aptr;
    int b = batch[i];
    float4 v = *(const float4*)(yx + (size_t)i * DD + c);
    v.x = v.x * scale[c + 0] + shift[c + 0]; v.x = v.x >= 0.f ? v.x : a * v.x;
    v.y = v.y * scale[c + 1] + shift[c + 1]; v.y = v.y >= 0.f ? v.y : a * v.y;
    v.z = v.z * scale[c + 2] + shift[c + 2]; v.z = v.z >= 0.f ? v.z : a * v.z;
    v.w = v.w * scale[c + 3] + shift[c + 3]; v.w = v.w >= 0.f ? v.w : a * v.w;
    *(float4*)(yx + (size_t)i * DD + c) = v;
    float* dst = gsumG + (size_t)b * DD + c;
    atomicAdd(dst + 0, v.x);
    atomicAdd(dst + 1, v.y);
    atomicAdd(dst + 2, v.z);
    atomicAdd(dst + 3, v.w);
    if ((g & 31) == 0) atomicAdd(&gcnt[b], 1.f);
}

// ---------------- GlobalModel: 32x256 @ 256x128, per-column BN (each thread owns a column) ----
__global__ __launch_bounds__(128) void global_model_kernel(const float* __restrict__ u,
                                                           const float* __restrict__ Gsum,
                                                           const float* __restrict__ Gcnt,
                                                           const float* __restrict__ Wg,
                                                           const float* __restrict__ gg,
                                                           const float* __restrict__ bg,
                                                           const float* __restrict__ ag,
                                                           float* __restrict__ uout, int Gn) {
    int c = threadIdx.x;           // 0..127
    float yv[32];
    for (int r = 0; r < Gn; ++r) {
        float invc = 1.f / fmaxf(Gcnt[r], 1.f);
        float acc = 0.f;
        for (int k = 0; k < DD; ++k)  acc += u[(size_t)r * DD + k] * Wg[(size_t)k * DD + c];
        for (int k = 0; k < DD; ++k)  acc += (Gsum[(size_t)r * DD + k] * invc) * Wg[(size_t)(DD + k) * DD + c];
        yv[r] = acc;
    }
    float m = 0.f;
    for (int r = 0; r < Gn; ++r) m += yv[r];
    m *= (1.f / (float)Gn);
    float var = 0.f;
    for (int r = 0; r < Gn; ++r) { float d = yv[r] - m; var += d * d; }
    var *= (1.f / (float)Gn);
    float sc = gg[c] * rsqrtf(var + 1e-5f);
    float sh = bg[c] - m * sc;
    float a = *ag;
    for (int r = 0; r < Gn; ++r) {
        float z = yv[r] * sc + sh;
        uout[(size_t)r * DD + c] = z >= 0.f ? z : a * z;
    }
}

// ---------------- host-side orchestration ----------------
extern "C" void kernel_launch(void* const* d_in, const int* in_sizes, int n_in,
                              void* d_out, int out_size, void* d_ws, size_t ws_size,
                              hipStream_t stream) {
    (void)n_in; (void)out_size; (void)ws_size;
    const float* x     = (const float*)d_in[0];
    const int*   eidx  = (const int*)d_in[1];
    const float* eattr = (const float*)d_in[2];
    const float* u     = (const float*)d_in[3];
    const int*   batch = (const int*)d_in[4];
    const float* We  = (const float*)d_in[5];
    const float* ge  = (const float*)d_in[6];
    const float* be  = (const float*)d_in[7];
    const float* ae  = (const float*)d_in[8];
    const float* Wn1 = (const float*)d_in[9];
    const float* gn1 = (const float*)d_in[10];
    const float* bn1 = (const float*)d_in[11];
    const float* an1 = (const float*)d_in[12];
    const float* Wn2 = (const float*)d_in[13];
    const float* gn2 = (const float*)d_in[14];
    const float* bn2 = (const float*)d_in[15];
    const float* an2 = (const float*)d_in[16];
    const float* Wg  = (const float*)d_in[17];
    const float* gg  = (const float*)d_in[18];
    const float* bg  = (const float*)d_in[19];
    const float* ag  = (const float*)d_in[20];

    const int N = in_sizes[0] / DD;
    const int E = in_sizes[1] / 2;
    const int G = in_sizes[3] / DD;
    const int* row = eidx;
    const int* col = eidx + E;

    float* xout = (float*)d_out;                    // N x D
    float* eout = xout + (size_t)N * DD;            // E x D
    float* uout = eout + (size_t)E * DD;            // G x D

    // workspace layout
    char* ws = (char*)d_ws;
    const size_t OFF_PWE   = 0;                                         // 384*128 bf16
    const size_t OFF_PWN1  = OFF_PWE  + (size_t)384 * DD * 2;           // 256*128 bf16
    const size_t OFF_PWN2  = OFF_PWN1 + (size_t)256 * DD * 2;
    const size_t OFF_ZERO  = OFF_PWN2 + (size_t)256 * DD * 2;
    const size_t OFF_STATS = OFF_ZERO;                                  // 3 sets x 512 floats
    const size_t OFF_CNT   = OFF_STATS + 8192;                          // N floats
    const size_t OFF_GSUM  = OFF_CNT + (((size_t)N * 4 + 255) & ~255ull);
    const size_t OFF_GCNT  = OFF_GSUM + (size_t)G * DD * 4;             // G floats
    const size_t OFF_AGG   = OFF_GCNT + 256;                            // N*D floats
    const size_t ZERO_END  = OFF_AGG + (size_t)N * DD * 4;
    const size_t OFF_YH    = (ZERO_END + 255) & ~255ull;                // E*D floats

    unsigned short* pWe  = (unsigned short*)(ws + OFF_PWE);
    unsigned short* pWn1 = (unsigned short*)(ws + OFF_PWN1);
    unsigned short* pWn2 = (unsigned short*)(ws + OFF_PWN2);
    float* st   = (float*)(ws + OFF_STATS);
    float* sum0 = st + 0 * 512, *sq0 = sum0 + 128, *sc0 = sum0 + 256, *sh0 = sum0 + 384;
    float* sum1 = st + 1 * 512, *sq1 = sum1 + 128, *sc1 = sum1 + 256, *sh1 = sum1 + 384;
    float* sum2 = st + 2 * 512, *sq2 = sum2 + 128, *sc2 = sum2 + 256, *sh2 = sum2 + 384;
    float* cntp  = (float*)(ws + OFF_CNT);
    float* gsumG = (float*)(ws + OFF_GSUM);
    float* gcnt  = (float*)(ws + OFF_GCNT);
    float* aggp  = (float*)(ws + OFF_AGG);
    float* yh    = (float*)(ws + OFF_YH);

    // 1) pack weights to bf16 fragment layout
    pack_w_kernel<<<(384 * DD + 255) / 256, 256, 0, stream>>>(We,  pWe,  384);
    pack_w_kernel<<<(256 * DD + 255) / 256, 256, 0, stream>>>(Wn1, pWn1, 256);
    pack_w_kernel<<<(256 * DD + 255) / 256, 256, 0, stream>>>(Wn2, pWn2, 256);

    // 2) zero accumulator region (stats, cnt, Gsum, Gcnt, agg)
    size_t zbytes = ZERO_END - OFF_ZERO;
    size_t z4 = zbytes / 16;
    zero_kernel<<<(unsigned)((z4 + 255) / 256), 256, 0, stream>>>((float4*)(ws + OFF_ZERO), z4);

    // 3) EdgeModel GEMM (K=384) -> y_e into eout region + stats
    gemm_bn_kernel<0, 12><<<(E + 63) / 64, 128, 0, stream>>>(
        x, row, col, eattr, u, batch, nullptr, nullptr, nullptr, pWe, eout, sum0, sq0, E);
    bn_finalize_kernel<<<1, 128, 0, stream>>>(sum0, sq0, ge, be, sc0, sh0, 1.f / (float)E);
    bn_apply_kernel<<<(unsigned)(((size_t)E * 32 + 255) / 256), 256, 0, stream>>>(
        eout, sc0, sh0, ae, (size_t)E * 32);

    // 4) NodeModel GEMM1 (K=256) -> y_h into ws + stats
    gemm_bn_kernel<1, 8><<<(E + 63) / 64, 128, 0, stream>>>(
        x, row, col, eattr, u, batch, eout, nullptr, nullptr, pWn1, yh, sum1, sq1, E);
    bn_finalize_kernel<<<1, 128, 0, stream>>>(sum1, sq1, gn1, bn1, sc1, sh1, 1.f / (float)E);
    h_scatter_kernel<<<(unsigned)(((size_t)E * 32 + 255) / 256), 256, 0, stream>>>(
        yh, sc1, sh1, an1, row, aggp, cntp, E);

    // 5) NodeModel GEMM2 (K=256) -> y_x into xout region + stats
    gemm_bn_kernel<2, 8><<<(N + 63) / 64, 128, 0, stream>>>(
        x, row, col, eattr, u, batch, eout, aggp, cntp, pWn2, xout, sum2, sq2, N);
    bn_finalize_kernel<<<1, 128, 0, stream>>>(sum2, sq2, gn2, bn2, sc2, sh2, 1.f / (float)N);
    x_scatter_kernel<<<(unsigned)(((size_t)N * 32 + 255) / 256), 256, 0, stream>>>(
        xout, sc2, sh2, an2, batch, gsumG, gcnt, N);

    // 6) GlobalModel (tiny): 32x256 @ 256x128 with per-column BN + PReLU
    global_model_kernel<<<1, 128, 0, stream>>>(u, gsumG, gcnt, Wg, gg, bg, ag, uout, G);
}
